// RNNModel_61916248539621
// MI455X (gfx1250) — matrix operand order repaired and checked by reading, not compiled, not run on hardware
//
#include <hip/hip_runtime.h>

// ---------------------------------------------------------------------------
// Fused 2-layer LSTM + dense head for MI455X (gfx1250, wave32, WMMA f16).
//
// Parallel decomposition: B=256 batch rows -> 16 workgroups x 16 rows.
// Each workgroup runs the full T=256 recurrence for both layers privately
// (h/c state in LDS+VGPRs, no inter-WG sync). Per step, per wave (16 waves):
//   layer1: [16x576]  @ [576 x 128cols]  -> 18 k-steps x 8 wmma tiles
//   layer2: [16x1024] @ [1024 x 128cols] -> 32 k-steps x 8 wmma tiles
// using v_wmma_f32_16x16x32_f16.
// ---------------------------------------------------------------------------

typedef __attribute__((ext_vector_type(16))) _Float16 v16h;
typedef __attribute__((ext_vector_type(8)))  _Float16 v8h;
typedef __attribute__((ext_vector_type(8)))  float    v8f;

#define BATCH 256
#define TSTEPS 256
#define FIN 64
#define UNITS 512
#define NTOT 2048            // 4*UNITS
#define KT1 18               // (64+512)/32 k-tiles, layer 1
#define KT2 32               // (512+512)/32 k-tiles, layer 2
#define S1  592              // LDS row stride (halves) for [x_t | h1], 576+pad
#define S2  1040             // LDS row stride (halves) for [h1 | h2], 1024+pad

// ---------------------------------------------------------------------------
// Weight pre-pack: f32 row-major [K x 2048] (Wa rows then Wb rows) -> f16
// WMMA B-fragments. Fragment (ntile, kt) = 32x16 (KxN) block, 512 halves,
// laid out so lane l owns 16 contiguous halves at frag_base + l*16:
//   n = ntile*16 + (l & 15),  k = kt*32 + (l>>4)*16 + j   (j = 0..15)
// ---------------------------------------------------------------------------
__global__ void pack_b_kernel(const float* __restrict__ Wa,
                              const float* __restrict__ Wb,
                              int rowsA, int KT, _Float16* __restrict__ dst,
                              int totalHalves) {
    int idx = blockIdx.x * blockDim.x + threadIdx.x;
    if (idx >= totalHalves) return;
    int j    = idx & 15;
    int lane = (idx >> 4) & 31;
    int frag = idx >> 9;
    int kt   = frag % KT;
    int nt   = frag / KT;
    int n = nt * 16 + (lane & 15);
    int k = kt * 32 + ((lane >> 4) << 4) + j;
    float v = (k < rowsA) ? Wa[(size_t)k * NTOT + n]
                          : Wb[(size_t)(k - rowsA) * NTOT + n];
    dst[idx] = (_Float16)v;
}

// A-fragment gather from LDS, contiguous-k 16-bit A 16x32 layout (same k order as B pack):
// lane m = l&15; halves 0-15 -> K = kb..kb+15,
// kb = (l>>4)*16.  Two 16B ds loads per lane.
__device__ __forceinline__ v16h loadA(const _Float16* __restrict__ S,
                                      int stride, int kt, int lane) {
    int m  = lane & 15;
    int kb = (lane >> 4) * 16;
    const _Float16* p = S + m * stride + kt * 32 + kb;
    v8h lo = *(const v8h*)(p);
    v8h hi = *(const v8h*)(p + 8);
    return __builtin_shufflevector(lo, hi, 0, 1, 2, 3, 4, 5, 6, 7,
                                   8, 9, 10, 11, 12, 13, 14, 15);
}

// One fused-gate GEMM: 8 accumulator tiles (4 gates x 2 n-subtiles of 16).
__device__ __forceinline__ void gemm8(const _Float16* __restrict__ S, int stride,
                                      int KT, const _Float16* __restrict__ pack,
                                      v8f acc[8], int w, int lane) {
#pragma unroll 2
    for (int kt = 0; kt < KT; ++kt) {
        v16h a = loadA(S, stride, kt, lane);
#pragma unroll
        for (int tI = 0; tI < 8; ++tI) {
            int g = tI >> 1, s = tI & 1;
            int ntile = g * 32 + w * 2 + s;            // (g*512 + w*32 + s*16)/16
            const v16h* bp = (const v16h*)(pack + (((size_t)ntile * KT + kt) << 9));
            v16h b = bp[lane];
            acc[tI] = __builtin_amdgcn_wmma_f32_16x16x32_f16(
                false, a, false, b, (short)0, acc[tI], false, false);
        }
    }
}

__device__ __forceinline__ float sigm(float x) {
    return 1.0f / (1.0f + __expf(-x));
}

// Gate activations + cell/hidden update in the WMMA C/D register layout:
// VGPR e holds row (e + (lane>>4)*8), column (lane&15) of each 16-wide tile.
__device__ __forceinline__ void lstm_update(const v8f acc[8], const float bias[8],
                                            v8f c[2],
                                            _Float16* __restrict__ hdst0, int stride0,
                                            _Float16* __restrict__ hdst1, int stride1,
                                            int w, int lane) {
    int rowoff = (lane >> 4) * 8;
    int nloc   = lane & 15;
#pragma unroll
    for (int s = 0; s < 2; ++s) {
        int col = w * 32 + s * 16 + nloc;
#pragma unroll
        for (int e = 0; e < 8; ++e) {
            float iv = sigm(acc[0 * 2 + s][e] + bias[0 * 2 + s]);
            float fv = sigm(acc[1 * 2 + s][e] + bias[1 * 2 + s]);
            float gv = fmaxf(acc[2 * 2 + s][e] + bias[2 * 2 + s], 0.0f);
            float ov = sigm(acc[3 * 2 + s][e] + bias[3 * 2 + s]);
            float cv = fv * c[s][e] + iv * gv;
            c[s][e] = cv;
            _Float16 hh = (_Float16)(ov * fmaxf(cv, 0.0f));
            int r = e + rowoff;
            hdst0[r * stride0 + col] = hh;
            if (hdst1) hdst1[r * stride1 + col] = hh;
        }
    }
}

__launch_bounds__(512, 1)
__global__ void lstm_fused_kernel(const float* __restrict__ x,
                                  const float* __restrict__ b1,
                                  const float* __restrict__ b2,
                                  const float* __restrict__ Wd,
                                  const float* __restrict__ bd,
                                  const _Float16* __restrict__ pack1,
                                  const _Float16* __restrict__ pack2,
                                  float* __restrict__ out) {
    __shared__ __align__(32) _Float16 sA1[16 * S1];  // [x_t(64) | h1(512)]
    __shared__ __align__(32) _Float16 sA2[16 * S2];  // [h1_t(512) | h2(512)]

    int tid  = threadIdx.x;
    int lane = tid & 31;
    int w    = tid >> 5;           // 16 waves: wave w owns units [w*32, w*32+32)
    int b0   = blockIdx.x * 16;

    // Zero initial hidden-state regions (h1 in sA1, h2 in sA2).
    for (int i = tid; i < 16 * 512; i += 512) {
        int r = i >> 9, cc = i & 511;
        sA1[r * S1 + 64  + cc] = (_Float16)0.0f;
        sA2[r * S2 + 512 + cc] = (_Float16)0.0f;
    }

    // Per-lane bias preload (bias depends only on the output column).
    float bi1[8], bi2[8];
#pragma unroll
    for (int tI = 0; tI < 8; ++tI) {
        int g = tI >> 1, s = tI & 1;
        int n = g * 512 + w * 32 + s * 16 + (lane & 15);
        bi1[tI] = b1[n];
        bi2[tI] = b2[n];
    }

    v8f c1[2] = {}, c2[2] = {};
    __syncthreads();

    for (int t = 0; t < TSTEPS; ++t) {
        // ---- stage x_t into sA1[:, 0:64] (f32 -> f16), prefetch t+1 ----
        {
            int r = tid >> 6, cc = tid & 63;
            size_t base0 = (size_t)(b0 + r)     * TSTEPS * FIN + (size_t)t * FIN + cc;
            size_t base1 = (size_t)(b0 + r + 8) * TSTEPS * FIN + (size_t)t * FIN + cc;
            sA1[r * S1 + cc]       = (_Float16)x[base0];
            sA1[(r + 8) * S1 + cc] = (_Float16)x[base1];
            if (t + 1 < TSTEPS) {
                __builtin_prefetch(&x[base0 + FIN], 0, 1);
                __builtin_prefetch(&x[base1 + FIN], 0, 1);
            }
        }
        __syncthreads();                       // x_t ready; prior-step writes visible

        // ---- layer 1 gates: [x_t | h1_{t-1}] @ [W1;U1] ----
        v8f acc[8] = {};
        gemm8(sA1, S1, KT1, pack1, acc, w, lane);
        __syncthreads();                       // all waves done reading sA1

        // h1_t -> sA1 h-region (next step) and sA2 first half (layer 2 input)
        lstm_update(acc, bi1, c1, sA1 + 64, S1, sA2, S2, w, lane);
        __syncthreads();                       // h1_t ready in sA2

        // ---- layer 2 gates: [h1_t | h2_{t-1}] @ [W2;U2] ----
        v8f acc2[8] = {};
        gemm8(sA2, S2, KT2, pack2, acc2, w, lane);
        __syncthreads();                       // all waves done reading sA2

        lstm_update(acc2, bi2, c2, sA2 + 512, S2, (_Float16*)nullptr, 0, w, lane);
    }
    __syncthreads();

    // ---- dense head: out[b] = sigmoid(h2_T . Wd + bd), one row per thread ----
    if (tid < 16) {
        float s = 0.0f;
        for (int j = 0; j < UNITS; ++j)
            s += (float)sA2[tid * S2 + 512 + j] * Wd[j];
        out[b0 + tid] = 1.0f / (1.0f + __expf(-(s + bd[0])));
    }
}

extern "C" void kernel_launch(void* const* d_in, const int* in_sizes, int n_in,
                              void* d_out, int out_size, void* d_ws, size_t ws_size,
                              hipStream_t stream) {
    const float* x  = (const float*)d_in[0];
    const float* W1 = (const float*)d_in[1];
    const float* U1 = (const float*)d_in[2];
    const float* b1 = (const float*)d_in[3];
    const float* W2 = (const float*)d_in[4];
    const float* U2 = (const float*)d_in[5];
    const float* b2 = (const float*)d_in[6];
    const float* Wd = (const float*)d_in[7];
    const float* bd = (const float*)d_in[8];

    _Float16* pack1 = (_Float16*)d_ws;                       // 128*18*512 halves
    const int tot1 = 128 * KT1 * 512;                        // 1,179,648
    const int tot2 = 128 * KT2 * 512;                        // 2,097,152
    _Float16* pack2 = pack1 + tot1;

    pack_b_kernel<<<(tot1 + 255) / 256, 256, 0, stream>>>(W1, U1, 64,  KT1, pack1, tot1);
    pack_b_kernel<<<(tot2 + 255) / 256, 256, 0, stream>>>(W2, U2, 512, KT2, pack2, tot2);

    lstm_fused_kernel<<<BATCH / 16, 512, 0, stream>>>(x, b1, b2, Wd, bd,
                                                      pack1, pack2, (float*)d_out);
}
